// RBM_89240830476533
// MI455X (gfx1250) — matrix operand order repaired and checked by reading, not compiled, not run on hardware
//
#include <hip/hip_runtime.h>
#include <math.h>

typedef __bf16 bf16;
typedef __attribute__((ext_vector_type(4)))  bf16  bf16x4;
typedef __attribute__((ext_vector_type(8)))  bf16  bf16x8;
typedef __attribute__((ext_vector_type(16))) bf16  bf16x16;
typedef __attribute__((ext_vector_type(8)))  float floatx8;

#define NUM_CLASSES 41
#define B_  8
#define C_  128
#define HW_ 16384   // 128*128
#define KTOT 2304   // 2C * 9
#define WROW 264    // padded window row stride (elements): 528B -> conflict-free frags

union FragBf { bf16x8 h[2]; bf16x16 v; };

// ---------------------------------------------------------------------------
// K1: pack conv weights OIHW f32 -> Wt[m][tap*256+cin] bf16 (K-major, tap outer)
// ---------------------------------------------------------------------------
__global__ void pack_w_kernel(const float* __restrict__ cw, bf16* __restrict__ Wt) {
    int i = blockIdx.x * 256 + threadIdx.x;         // 128*2304 total
    if (i >= 128 * KTOT) return;
    int m   = i / KTOT;
    int k   = i - m * KTOT;
    int tap = k >> 8;                               // k / 256
    int cin = k & 255;
    Wt[(size_t)m * KTOT + k] = (bf16)cw[((size_t)m * 256 + cin) * 9 + tap];
}

// ---------------------------------------------------------------------------
// K2: build X = concat(r+d, r*sigmoid(d)) as bf16 NHWC: Xt[b][p][cin 0..255]
// grid = B*512 blocks (32 px each), 256 threads
// ---------------------------------------------------------------------------
__global__ void build_x_kernel(const float* __restrict__ r, const float* __restrict__ d,
                               bf16* __restrict__ Xt) {
    const int blk = blockIdx.x;
    const int b   = blk >> 9;
    const int p0  = (blk & 511) * 32;
    const int pi  = threadIdx.x & 31;
    const int cg  = threadIdx.x >> 5;               // 0..7
    const int p   = p0 + pi;
    const size_t baseP = ((size_t)b * HW_ + p) * 256;
    #pragma unroll 4
    for (int cc = 0; cc < 32; ++cc) {
        int c  = cg * 32 + cc;                      // 0..255
        int ci = c & 127;
        size_t ri = ((size_t)b * C_ + ci) * HW_ + p; // coalesced across lanes (p)
        float rv = r[ri];
        float dv = d[ri];
        float v  = (c < 128) ? (rv + dv) : (rv * (1.0f / (1.0f + expf(-dv))));
        Xt[baseP + c] = (bf16)v;
    }
}

// ---------------------------------------------------------------------------
// K3: implicit-GEMM 3x3 conv via WMMA bf16 -> f32.
// Block: 8 waves in a 4(M) x 2(N) grid; block tile M=128 x N=64 px.
// Each wave owns a 32x32 output tile: per K-step 2 A-frags + 2 B-frags -> 4 WMMAs
// (1 ds_load_b128 per WMMA). Halo window staged to LDS once (1 barrier).
// grid = B*256; out = fuse, f32 NCHW (written into d_out).
// ---------------------------------------------------------------------------
__launch_bounds__(256, 2)
__global__ void conv_wmma_kernel(const bf16* __restrict__ Xt,
                                 const bf16* __restrict__ Wt,
                                 float* __restrict__ out) {
    // window: rows y-1..y+1, cols x0-1..x0+64, 256 ch; padded row stride 264
    __shared__ __align__(16) bf16 Win[3][66][WROW];   // 104,544 B

    const int blk = blockIdx.x;
    const int b   = blk >> 8;                       // 256 tiles per batch
    const int pt  = blk & 255;
    const int y   = pt >> 1;                        // 2 tiles of 64 px per row
    const int x0  = (pt & 1) * 64;

    const int tid   = threadIdx.x;
    const int wid   = tid >> 5;
    const int wm    = wid & 3;                      // M group: out-ch 32*wm..+31
    const int wn    = wid >> 2;                     // N group: px   32*wn..+31
    const int lane  = tid & 31;
    const int lr    = lane & 15;
    const int half  = lane >> 4;
    const int koffA = half * 8;                     // A: K interleave granularity 8
    const int koffB = half * 16;                    // unused: B loads use koffA (granularity 8)

    // ---- stage halo window (zero-padded) with b128 chunks, one barrier ----
    const int NCHUNK = 3 * 66 * 32;                 // 6336 16B chunks
    for (int ci = tid; ci < NCHUNK; ci += 256) {
        int row = ci / (66 * 32);
        int rem = ci - row * (66 * 32);
        int col = rem >> 5;
        int cg  = rem & 31;
        int yy  = y + row - 1;
        int xx  = x0 + col - 1;
        bf16x8 v = {};
        if (yy >= 0 && yy < 128 && xx >= 0 && xx < 128)
            v = *(const bf16x8*)(Xt + (((size_t)b * HW_ + (size_t)yy * 128 + xx) * 256) + cg * 8);
        *(bf16x8*)(&Win[row][col][cg * 8]) = v;
    }
    __syncthreads();

    floatx8 acc00 = {}, acc01 = {}, acc10 = {}, acc11 = {};  // [Mtile][Ntile]
    const bf16* wrow0 = Wt + (size_t)(wm * 32 + lr) * KTOT;
    const bf16* wrow1 = wrow0 + (size_t)16 * KTOT;

    for (int tap = 0; tap < 9; ++tap) {
        const int dy = tap / 3;                     // window row 0..2
        const int dx = tap % 3;                     // window col shift 0..2
        #pragma unroll
        for (int c0 = 0; c0 < 256; c0 += 32) {
            const int k0 = tap * 256 + c0;
            FragBf a0, a1, b0, b1;
            a0.h[0] = *(const bf16x8*)(wrow0 + k0 + koffA);
            a0.h[1] = *(const bf16x8*)(wrow0 + k0 + 16 + koffA);
            a1.h[0] = *(const bf16x8*)(wrow1 + k0 + koffA);
            a1.h[1] = *(const bf16x8*)(wrow1 + k0 + 16 + koffA);

            const bf16* brow = &Win[dy][wn * 32 + lr + dx][c0 + koffA];
            b0.h[0] = *(const bf16x8*)(brow);
            b0.h[1] = *(const bf16x8*)(brow + 16);
            b1.h[0] = *(const bf16x8*)(brow + 16 * WROW);
            b1.h[1] = *(const bf16x8*)(brow + 16 * WROW + 16);

            acc00 = __builtin_amdgcn_wmma_f32_16x16x32_bf16(
                false, a0.v, false, b0.v, (short)0, acc00, false, false);
            acc01 = __builtin_amdgcn_wmma_f32_16x16x32_bf16(
                false, a0.v, false, b1.v, (short)0, acc01, false, false);
            acc10 = __builtin_amdgcn_wmma_f32_16x16x32_bf16(
                false, a1.v, false, b0.v, (short)0, acc10, false, false);
            acc11 = __builtin_amdgcn_wmma_f32_16x16x32_bf16(
                false, a1.v, false, b1.v, (short)0, acc11, false, false);
        }
    }

    // D layout: lanes 0-15 -> M = vgpr i, N = lane; lanes 16-31 -> M = 8+i, N = lane-16
    const int xcol = x0 + wn * 32 + lr;
    float* op = out + (((size_t)b * C_ + wm * 32 + half * 8) * HW_ + (size_t)y * 128 + xcol);
    #pragma unroll
    for (int i = 0; i < 8; ++i) {
        op[(size_t)i * HW_]              = acc00[i];
        op[(size_t)i * HW_ + 16]         = acc01[i];
        op[(size_t)(i + 16) * HW_]       = acc10[i];
        op[(size_t)(i + 16) * HW_ + 16]  = acc11[i];
    }
}

// ---------------------------------------------------------------------------
// K4: per-class sums + counts via LDS f32 atomics. grid = B*16, 256 threads.
// in: f32 NCHW. label: [B,512,512], nearest-resize = stride-4 gather.
// ---------------------------------------------------------------------------
__global__ void seg_sums_kernel(const float* __restrict__ in, const int* __restrict__ label,
                                float* __restrict__ gsums, float* __restrict__ gcnt) {
    __shared__ float lsum[NUM_CLASSES * C_];
    __shared__ float lcnt[NUM_CLASSES];
    const int b    = blockIdx.x >> 4;
    const int slab = blockIdx.x & 15;
    const int tid  = threadIdx.x;
    for (int i = tid; i < NUM_CLASSES * C_; i += 256) lsum[i] = 0.0f;
    if (tid < NUM_CLASSES) lcnt[tid] = 0.0f;
    __syncthreads();

    for (int it = 0; it < 4; ++it) {
        int p  = slab * 1024 + it * 256 + tid;      // coalesced pixel access
        int yy = p >> 7, xx = p & 127;
        int cls = label[(size_t)b * 512 * 512 + (size_t)(yy * 4) * 512 + xx * 4];
        atomicAdd(&lcnt[cls], 1.0f);
        const float* src = in + (size_t)b * C_ * HW_ + p;
        #pragma unroll 4
        for (int ch = 0; ch < C_; ++ch)
            atomicAdd(&lsum[cls * C_ + ch], src[(size_t)ch * HW_]);
    }
    __syncthreads();
    for (int i = tid; i < NUM_CLASSES * C_; i += 256)
        atomicAdd(&gsums[(size_t)b * NUM_CLASSES * C_ + i], lsum[i]);
    if (tid < NUM_CLASSES)
        atomicAdd(&gcnt[b * NUM_CLASSES + tid], lcnt[tid]);
}

// ---------------------------------------------------------------------------
// K5: gate finalize: mean/where -> L2-normalize over classes -> sum -> MLP -> sigmoid
// grid = B, block = 128 (thread = channel)
// ---------------------------------------------------------------------------
__global__ void gate_kernel(const float* __restrict__ gsums, const float* __restrict__ gcnt,
                            const float* __restrict__ base, const float* __restrict__ w1,
                            const float* __restrict__ w2, float* __restrict__ g) {
    __shared__ float ls[C_];
    __shared__ float lh[8];
    const int b = blockIdx.x;
    const int c = threadIdx.x;
    float ssum = 0.0f, ssq = 0.0f;
    for (int l = 0; l < NUM_CLASSES; ++l) {
        float cnt = gcnt[b * NUM_CLASSES + l];
        float v   = (cnt > 0.0f)
                  ? gsums[((size_t)b * NUM_CLASSES + l) * C_ + c] / fmaxf(cnt, 1.0f)
                  : base[((size_t)b * NUM_CLASSES + l) * C_ + c];
        ssq  += v * v;
        ssum += v;
    }
    float s = ssum / fmaxf(sqrtf(ssq), 1e-12f);     // (sum over classes) / max(norm, eps)
    ls[c] = s;
    __syncthreads();
    if (c < 8) {
        float h = 0.0f;
        for (int k = 0; k < C_; ++k) h += w1[c * C_ + k] * ls[k];
        lh[c] = fmaxf(h, 0.0f);
    }
    __syncthreads();
    float o = 0.0f;
    #pragma unroll
    for (int k = 0; k < 8; ++k) o += w2[c * 8 + k] * lh[k];
    g[b * C_ + c] = 1.0f / (1.0f + expf(-o));
}

// ---------------------------------------------------------------------------
// K6: out = fuse*g_f + d*g_d  (fuse already in out; float4 in-place)
// ---------------------------------------------------------------------------
__global__ void final_kernel(const float* __restrict__ d_in, const float* __restrict__ gf,
                             const float* __restrict__ gd, float* __restrict__ out, int n4) {
    int i = blockIdx.x * blockDim.x + threadIdx.x;
    if (i >= n4) return;
    size_t e  = (size_t)i * 4;
    int bc = (int)(e >> 14);                        // (b*128 + c)
    int c  = bc & 127;
    int b  = bc >> 7;
    float a1 = gf[b * C_ + c];
    float a2 = gd[b * C_ + c];
    float4 f  = ((const float4*)out)[i];
    float4 dd = ((const float4*)d_in)[i];
    f.x = f.x * a1 + dd.x * a2;
    f.y = f.y * a1 + dd.y * a2;
    f.z = f.z * a1 + dd.z * a2;
    f.w = f.w * a1 + dd.w * a2;
    ((float4*)out)[i] = f;
}

// ---------------------------------------------------------------------------
extern "C" void kernel_launch(void* const* d_in, const int* in_sizes, int n_in,
                              void* d_out, int out_size, void* d_ws, size_t ws_size,
                              hipStream_t stream) {
    const float* r      = (const float*)d_in[0];
    const float* dten   = (const float*)d_in[1];
    const int*   label  = (const int*)d_in[2];
    const float* conv_w = (const float*)d_in[3];
    const float* f_w1   = (const float*)d_in[4];
    const float* f_w2   = (const float*)d_in[5];
    const float* d_w1   = (const float*)d_in[6];
    const float* d_w2   = (const float*)d_in[7];
    const float* base_f = (const float*)d_in[8];
    const float* base_d = (const float*)d_in[9];
    float* out = (float*)d_out;
    char*  ws  = (char*)d_ws;

    // workspace layout
    const size_t OFF_W     = (size_t)B_ * HW_ * 256 * sizeof(bf16);   // 64 MB im2col src
    const size_t OFF_STATS = OFF_W + (1u << 20);                       // 1 MB weight slot
    const size_t N_SUM     = (size_t)B_ * NUM_CLASSES * C_;            // 41984
    const size_t N_CNT     = (size_t)B_ * NUM_CLASSES;                 // 328
    bf16*  Xt     = (bf16*)ws;
    bf16*  Wt     = (bf16*)(ws + OFF_W);
    float* sums_d = (float*)(ws + OFF_STATS);
    float* cnt_d  = sums_d + N_SUM;
    float* sums_f = cnt_d + N_CNT;
    float* cnt_f  = sums_f + N_SUM;
    float* g_d    = cnt_f + N_CNT;
    float* g_f    = g_d + B_ * C_;

    // zero atomic accumulators every call (memset node is graph-capturable)
    hipMemsetAsync(sums_d, 0, 2 * (N_SUM + N_CNT) * sizeof(float), stream);

    pack_w_kernel<<<(128 * KTOT + 255) / 256, 256, 0, stream>>>(conv_w, Wt);
    build_x_kernel<<<B_ * 512, 256, 0, stream>>>(r, dten, Xt);

    // gate for d (independent of conv)
    seg_sums_kernel<<<B_ * 16, 256, 0, stream>>>(dten, label, sums_d, cnt_d);
    gate_kernel<<<B_, 128, 0, stream>>>(sums_d, cnt_d, base_d, d_w1, d_w2, g_d);

    // fuse = conv3x3(concat(add,mul)) -> deposited directly into d_out
    conv_wmma_kernel<<<B_ * 256, 256, 0, stream>>>(Xt, Wt, out);

    // gate for fuse
    seg_sums_kernel<<<B_ * 16, 256, 0, stream>>>(out, label, sums_f, cnt_f);
    gate_kernel<<<B_, 128, 0, stream>>>(sums_f, cnt_f, base_f, f_w1, f_w2, g_f);

    // out = fuse*g_f + d*g_d
    final_kernel<<<(B_ * C_ * HW_ / 4 + 255) / 256, 256, 0, stream>>>(
        dten, g_f, g_d, out, B_ * C_ * HW_ / 4);
}
